// StateSpace_489626272327
// MI455X (gfx1250) — matrix operand order repaired and checked by reading, not compiled, not run on hardware
//
#include <hip/hip_runtime.h>
#include <math.h>

// ---------------------------------------------------------------------------
// State-space scan  h_{t+1} = A h_t + B u_t,  out = tanh(h_t) (pre-update)
// ORDER=2048, IN_DIM=64, T=4096.
//
// Chunked-scan decomposition (L=64 chunk length, C=64 chunks):
//   phase 1: batched local recurrence (zero init) -> 63x GEMM 2048x2048x64
//   phase 2: s_{c+1} = A^64 s_c + q_c            -> 63 sequential matvecs
//            (A^64 via 6 WMMA GEMM squarings)
//   phase 3: batched recurrence from true s_c    -> 63x GEMM + 64 emits
// All GEMMs: bf16 operands, f32 accumulate, v_wmma_f32_16x16x32_bf16.
// Workspace use: ~87 MB of d_ws.
// ---------------------------------------------------------------------------

#define ORDER   2048
#define IN_DIM  64
#define SEQ     4096
static constexpr int CHUNK_LEN = 64;   // L
static constexpr int NCHUNK    = 64;   // C = SEQ / L

typedef __bf16 bf16_t;
typedef __attribute__((ext_vector_type(16))) __bf16 v16bf;
typedef __attribute__((ext_vector_type(8)))  __bf16 v8bf;
typedef __attribute__((ext_vector_type(8)))  float  v8f;

static __device__ __forceinline__ v16bf cat8(v8bf lo, v8bf hi) {
  return __builtin_shufflevector(lo, hi, 0,1,2,3,4,5,6,7,8,9,10,11,12,13,14,15);
}

// A-operand fragment, 16x32 bf16, row-major lda (K contiguous).
// ISA layout: lanes 0-15 (M=lane): K {0..7,16..23}; lanes 16-31: K {8..15,24..31}.
static __device__ __forceinline__ v16bf
load_a_frag(const bf16_t* __restrict__ A, int m0, int k0, int lda, int lane) {
  int half = lane >> 4;
  const bf16_t* base = A + (size_t)(m0 + (lane & 15)) * lda + k0 + half * 8;
  v8bf lo = *(const v8bf*)(base);        // K +0..7   (or +8..15)
  v8bf hi = *(const v8bf*)(base + 16);   // K +16..23 (or +24..31)
  return cat8(lo, hi);
}

// B-operand fragment, 32x16 bf16 from [n][k] storage (K contiguous per column).
// ISA layout: lanes 0-15 (N=lane): K {0..7,16..23}; lanes 16-31: K {8..15,24..31}.
static __device__ __forceinline__ v16bf
load_b_frag(const bf16_t* __restrict__ Bt, int n0, int k0, int ldb, int lane) {
  int half = lane >> 4;
  const bf16_t* base = Bt + (size_t)(n0 + (lane & 15)) * ldb + k0 + half * 8;
  v8bf lo = *(const v8bf*)(base);
  v8bf hi = *(const v8bf*)(base + 16);
  return cat8(lo, hi);
}

// ---------------------------------------------------------------------------
// One batched recurrence step:  Hout[c][:] = A @ Hin[c][:] + BU[:, c*L + jcol]
// Hin/Hout stored [NCHUNK][ORDER] (state per chunk contiguous => B-operand layout).
// Grid: 64 blocks (32 rows each), block = 128 threads = 4 waves.
// Wave w: rowblock (w&1), column half (w>>1) -> two 16x16 f32 accumulators.
// B tile (64 chunks x 256 K) staged in padded LDS.
// ---------------------------------------------------------------------------
__global__ __launch_bounds__(128) void gemm_step_kernel(
    const bf16_t* __restrict__ A,     // [ORDER][ORDER]
    const bf16_t* __restrict__ Hin,   // [NCHUNK][ORDER]
    const float*  __restrict__ BU,    // [ORDER][SEQ]
    int jcol,
    float*  __restrict__ HfOut,       // [NCHUNK][ORDER]
    bf16_t* __restrict__ HbfOut)      // [NCHUNK][ORDER]
{
  constexpr int KC  = 256;
  constexpr int LDB = KC + 8;         // pad: stride 528B -> banks 4 apart
  __shared__ bf16_t sB[NCHUNK][LDB];

  const int tid  = threadIdx.x;
  const int lane = tid & 31;
  const int w    = tid >> 5;
  const int rb   = w & 1;
  const int ch   = w >> 1;
  const int m0   = blockIdx.x * 32 + rb * 16;

  v8f acc0 = {}, acc1 = {};

  for (int kc = 0; kc < ORDER; kc += KC) {
    // stage Hin[0..63][kc..kc+255] -> LDS (coalesced global, conflict-free ds)
    #pragma unroll
    for (int i = 0; i < 16; ++i) {
      int u  = i * 128 + tid;          // 2048 v8bf units
      int c  = u >> 5;
      int ko = (u & 31) * 8;
      *(v8bf*)&sB[c][ko] = *(const v8bf*)(Hin + (size_t)c * ORDER + kc + ko);
    }
    __syncthreads();

    if (kc + KC < ORDER)
      __builtin_prefetch(A + (size_t)m0 * ORDER + kc + KC, 0, 0);

    #pragma unroll
    for (int kk = 0; kk < KC; kk += 32) {
      v16bf a  = load_a_frag(A, m0, kc + kk, ORDER, lane);
      v16bf b0 = load_b_frag(&sB[0][0], (2 * ch + 0) * 16, kk, LDB, lane);
      v16bf b1 = load_b_frag(&sB[0][0], (2 * ch + 1) * 16, kk, LDB, lane);
      acc0 = __builtin_amdgcn_wmma_f32_16x16x32_bf16(false, a, false, b0,
                                                     (short)0, acc0, false, false);
      acc1 = __builtin_amdgcn_wmma_f32_16x16x32_bf16(false, a, false, b1,
                                                     (short)0, acc1, false, false);
    }
    __syncthreads();
  }

  // epilogue: add BU column, write f32 + bf16 state
  const int half = lane >> 4;
  #pragma unroll
  for (int i = 0; i < 2; ++i) {
    v8f acc = i ? acc1 : acc0;
    int c = (2 * ch + i) * 16 + (lane & 15);
    #pragma unroll
    for (int v = 0; v < 8; ++v) {
      int m = m0 + v + 8 * half;
      float x = acc[v] + BU[(size_t)m * SEQ + c * CHUNK_LEN + jcol];
      HfOut[(size_t)c * ORDER + m]  = x;
      HbfOut[(size_t)c * ORDER + m] = (bf16_t)x;
    }
  }
}

// ---------------------------------------------------------------------------
// Matrix squaring  Q = P @ P  (2048^3, bf16 in, f32 acc), writing Q row-major
// and Q^T (both bf16) so the result can feed the next squaring's B operand.
// Grid (16,16); block 256 = 8 waves; WG tile 128x128; B tile staged in LDS.
// ---------------------------------------------------------------------------
__global__ __launch_bounds__(256) void gemm_sq_kernel(
    const bf16_t* __restrict__ P,    // [ORDER][ORDER] row-major
    const bf16_t* __restrict__ Pt,   // [ORDER][ORDER] = P^T
    bf16_t* __restrict__ Q,          // row-major result
    bf16_t* __restrict__ Qt)         // transposed result
{
  constexpr int KC  = 128;
  constexpr int LDB = KC + 8;
  __shared__ bf16_t sB[128][LDB];

  const int tid   = threadIdx.x;
  const int lane  = tid & 31;
  const int w     = tid >> 5;                 // 0..7 -> rowblock
  const int m0    = blockIdx.x * 128 + w * 16;
  const int nbase = blockIdx.y * 128;

  v8f acc[8] = {};

  for (int kc = 0; kc < ORDER; kc += KC) {
    #pragma unroll
    for (int i = 0; i < 8; ++i) {
      int u  = i * 256 + tid;                 // 2048 v8bf units
      int n  = u >> 4;                        // 16 units per 128-elem row
      int ko = (u & 15) * 8;
      *(v8bf*)&sB[n][ko] = *(const v8bf*)(Pt + (size_t)(nbase + n) * ORDER + kc + ko);
    }
    __syncthreads();

    if (kc + KC < ORDER)
      __builtin_prefetch(P + (size_t)m0 * ORDER + kc + KC, 0, 0);

    #pragma unroll
    for (int kk = 0; kk < KC; kk += 32) {
      v16bf a = load_a_frag(P, m0, kc + kk, ORDER, lane);
      #pragma unroll
      for (int nb = 0; nb < 8; ++nb) {
        v16bf b = load_b_frag(&sB[0][0], nb * 16, kk, LDB, lane);
        acc[nb] = __builtin_amdgcn_wmma_f32_16x16x32_bf16(false, a, false, b,
                                                          (short)0, acc[nb], false, false);
      }
    }
    __syncthreads();
  }

  const int half = lane >> 4;
  #pragma unroll
  for (int nb = 0; nb < 8; ++nb) {
    int n = nbase + nb * 16 + (lane & 15);
    #pragma unroll
    for (int v = 0; v < 8; ++v) {
      int m = m0 + v + 8 * half;
      bf16_t x = (bf16_t)acc[nb][v];
      Q [(size_t)m * ORDER + n] = x;
      Qt[(size_t)n * ORDER + m] = x;
    }
  }
}

// ---------------------------------------------------------------------------
// Phase-2 matvec: y = M @ x + q   (M bf16 2048x2048). One wave per row.
// ---------------------------------------------------------------------------
__global__ __launch_bounds__(256) void matvec_kernel(
    const bf16_t* __restrict__ M, const float* __restrict__ x,
    const float* __restrict__ q, float* __restrict__ y)
{
  const int w = threadIdx.x >> 5, lane = threadIdx.x & 31;
  const int row = blockIdx.x * 8 + w;
  const bf16_t* mr = M + (size_t)row * ORDER;
  float sum = 0.f;
  for (int k = lane; k < ORDER; k += 32) sum += (float)mr[k] * x[k];
  for (int off = 16; off; off >>= 1) sum += __shfl_xor(sum, off, 32);
  if (lane == 0) y[row] = sum + q[row];
}

// BU = weight_b @ u : [ORDER][SEQ] f32
__global__ __launch_bounds__(256) void bu_kernel(
    const float* __restrict__ Bw, const float* __restrict__ U,
    float* __restrict__ BU)
{
  int t = blockIdx.x * 256 + threadIdx.x;
  int m = blockIdx.y;
  float acc = 0.f;
  #pragma unroll
  for (int k = 0; k < IN_DIM; ++k) acc += Bw[m * IN_DIM + k] * U[(size_t)k * SEQ + t];
  BU[(size_t)m * SEQ + t] = acc;
}

// A f32 -> bf16 row-major + transposed (one-time)
__global__ __launch_bounds__(256) void cvt_dual_kernel(
    const float* __restrict__ A, bf16_t* __restrict__ Ab, bf16_t* __restrict__ Abt)
{
  size_t id = (size_t)blockIdx.x * 256 + threadIdx.x;
  int m = (int)(id >> 11), k = (int)(id & 2047);
  bf16_t v = (bf16_t)A[id];
  Ab[id] = v;
  Abt[(size_t)k * ORDER + m] = v;
}

// f32 -> bf16 (states)
__global__ __launch_bounds__(256) void cvt_bf16_kernel(
    const float* __restrict__ src, bf16_t* __restrict__ dst)
{
  size_t id = (size_t)blockIdx.x * 256 + threadIdx.x;
  dst[id] = (bf16_t)src[id];
}

// phase-1 init: h^loc_1 = bu_{c*L}  (h^loc_0 = 0)
__global__ __launch_bounds__(256) void init_h1_kernel(
    const float* __restrict__ BU, float* __restrict__ Hf, bf16_t* __restrict__ Hbf)
{
  int id = blockIdx.x * 256 + threadIdx.x;   // NCHUNK*ORDER
  int c = id >> 11, m = id & 2047;
  float v = BU[(size_t)m * SEQ + c * CHUNK_LEN];
  Hf[id] = v; Hbf[id] = (bf16_t)v;
}

// s_0 = ones
__global__ __launch_bounds__(256) void init_s0_kernel(float* __restrict__ Sf)
{
  int id = blockIdx.x * 256 + threadIdx.x;
  if (id < ORDER) Sf[id] = 1.0f;
}

// out[:, c*L + j] = tanh(h)   (pre-update state)
__global__ __launch_bounds__(256) void emit_kernel(
    const float* __restrict__ Hf, int jcol, float* __restrict__ out)
{
  int id = blockIdx.x * 256 + threadIdx.x;   // NCHUNK*ORDER
  int c = id >> 11, m = id & 2047;
  out[(size_t)m * SEQ + c * CHUNK_LEN + jcol] = tanhf(Hf[id]);
}

// ---------------------------------------------------------------------------
extern "C" void kernel_launch(void* const* d_in, const int* in_sizes, int n_in,
                              void* d_out, int out_size, void* d_ws, size_t ws_size,
                              hipStream_t stream) {
  (void)in_sizes; (void)n_in; (void)out_size; (void)ws_size;
  const float* dU  = (const float*)d_in[0];  // [64][4096]
  const float* dA  = (const float*)d_in[1];  // [2048][2048]
  const float* dBw = (const float*)d_in[2];  // [2048][64]
  float* dOut = (float*)d_out;               // [2048][4096]

  // workspace carve-up (~87 MB total)
  char*  ws  = (char*)d_ws;
  size_t off = 0;
  auto alloc = [&](size_t bytes) -> void* {
    void* p = ws + off;
    off += (bytes + 255) & ~(size_t)255;
    return p;
  };
  const size_t MAT_BF = (size_t)ORDER * ORDER * sizeof(bf16_t);
  const size_t ST_F   = (size_t)NCHUNK * ORDER * sizeof(float);
  const size_t ST_BF  = (size_t)NCHUNK * ORDER * sizeof(bf16_t);

  float*  BU   = (float*)alloc((size_t)ORDER * SEQ * sizeof(float));
  bf16_t* Abf  = (bf16_t*)alloc(MAT_BF);
  bf16_t* AbfT = (bf16_t*)alloc(MAT_BF);
  bf16_t* Pa   = (bf16_t*)alloc(MAT_BF);
  bf16_t* PaT  = (bf16_t*)alloc(MAT_BF);
  bf16_t* Pb   = (bf16_t*)alloc(MAT_BF);
  bf16_t* PbT  = (bf16_t*)alloc(MAT_BF);
  float*  Hf0  = (float*)alloc(ST_F);
  float*  Hf1  = (float*)alloc(ST_F);
  bf16_t* Hb0  = (bf16_t*)alloc(ST_BF);
  bf16_t* Hb1  = (bf16_t*)alloc(ST_BF);
  float*  Sf   = (float*)alloc(ST_F);
  bf16_t* Sbf  = (bf16_t*)alloc(ST_BF);

  float*  Hf[2]  = {Hf0, Hf1};
  bf16_t* Hbf[2] = {Hb0, Hb1};

  // 0) precision prep + BU = B @ u
  cvt_dual_kernel<<<(ORDER * ORDER) / 256, 256, 0, stream>>>(dA, Abf, AbfT);
  bu_kernel<<<dim3(SEQ / 256, ORDER), 256, 0, stream>>>(dBw, dU, BU);

  // 1) A^64 by repeated squaring (A^2, A^4, ..., A^64): 6 WMMA GEMMs
  const bf16_t* Pcur = Abf;  const bf16_t* PcurT = AbfT;
  bf16_t* ping[2][2] = {{Pa, PaT}, {Pb, PbT}};
  for (int s = 0; s < 6; ++s) {
    bf16_t* Qm = ping[s & 1][0];  bf16_t* QmT = ping[s & 1][1];
    gemm_sq_kernel<<<dim3(16, 16), 256, 0, stream>>>(Pcur, PcurT, Qm, QmT);
    Pcur = Qm;  PcurT = QmT;
  }
  const bf16_t* A64 = Pcur;   // row-major bf16

  // 2) phase 1: batched local recurrence over 64 chunks (zero init)
  init_h1_kernel<<<(NCHUNK * ORDER) / 256, 256, 0, stream>>>(BU, Hf[0], Hbf[0]);
  int cur = 0;
  for (int j = 1; j < CHUNK_LEN; ++j) {
    gemm_step_kernel<<<ORDER / 32, 128, 0, stream>>>(Abf, Hbf[cur], BU, j,
                                                     Hf[cur ^ 1], Hbf[cur ^ 1]);
    cur ^= 1;
  }
  const float* qloc = Hf[cur];   // q_c = h^loc_{c,L}, laid out [c][ORDER]

  // 3) phase 2: s_{c+1} = A^64 s_c + q_c  (63 sequential matvecs)
  init_s0_kernel<<<ORDER / 256, 256, 0, stream>>>(Sf);
  for (int c = 0; c + 1 < NCHUNK; ++c)
    matvec_kernel<<<ORDER / 8, 256, 0, stream>>>(A64,
        Sf + (size_t)c * ORDER, qloc + (size_t)c * ORDER,
        Sf + (size_t)(c + 1) * ORDER);
  cvt_bf16_kernel<<<(NCHUNK * ORDER) / 256, 256, 0, stream>>>(Sf, Sbf);

  // 4) phase 3: batched recurrence from true chunk states, emitting tanh(h_t)
  const float*  curF = Sf;
  const bf16_t* curB = Sbf;
  int nb = 0;
  for (int j = 0; j < CHUNK_LEN; ++j) {
    emit_kernel<<<(NCHUNK * ORDER) / 256, 256, 0, stream>>>(curF, j, dOut);
    if (j + 1 < CHUNK_LEN) {
      gemm_step_kernel<<<ORDER / 32, 128, 0, stream>>>(Abf, curB, BU, j,
                                                       Hf[nb], Hbf[nb]);
      curF = Hf[nb];  curB = Hbf[nb];  nb ^= 1;
    }
  }
}
